// EnsembleLRULayer_67980742362005
// MI455X (gfx1250) — matrix operand order repaired and checked by reading, not compiled, not run on hardware
//
#include <hip/hip_runtime.h>
#include <hip/hip_bf16.h>
#include <math.h>

typedef _Float16 f16;
typedef _Float16 v16h __attribute__((ext_vector_type(16)));
typedef _Float16 v8h  __attribute__((ext_vector_type(8)));
typedef float    v8f  __attribute__((ext_vector_type(8)));

#define E_ 4
#define D_ 256
#define B_ 16
#define T_ 1024
#define MT 16384            // B*T
#define LN_EPS 1e-5f

// ---------------------------------------------------------------------------
// Small prep kernels
// ---------------------------------------------------------------------------
__global__ void prep_params_kernel(const float* __restrict__ params_log,
                                   float* __restrict__ frb, float* __restrict__ fib,
                                   float* __restrict__ gmb) {
  int i = blockIdx.x * blockDim.x + threadIdx.x;
  if (i >= E_ * D_) return;
  float nu    = expf(params_log[i]);
  float theta = expf(params_log[E_ * D_ + i]);
  float gamma = expf(params_log[2 * E_ * D_ + i]);
  float mag   = expf(-nu);
  frb[i] = mag * cosf(theta);
  fib[i] = mag * sinf(theta);
  gmb[i] = gamma;
}

__global__ void cvt_f32_f16_kernel(const float* __restrict__ src,
                                   f16* __restrict__ dst, int n) {
  int i = blockIdx.x * blockDim.x + threadIdx.x;
  int stride = gridDim.x * blockDim.x;
  for (; i < n; i += stride) dst[i] = (f16)src[i];
}

// dst[mat][n][dstKoff + k] (row length dstLdk) = scale * src[mat][k][n]
__global__ void transpose_cvt_kernel(const float* __restrict__ src,
                                     f16* __restrict__ dst,
                                     int K, int N, int nmat,
                                     int dstLdk, int dstKoff, float scale) {
  long total = (long)nmat * K * N;
  long i = (long)blockIdx.x * blockDim.x + threadIdx.x;
  long stride = (long)gridDim.x * blockDim.x;
  for (; i < total; i += stride) {
    int mat = (int)(i / ((long)K * N));
    int rem = (int)(i - (long)mat * K * N);
    int k = rem / N;
    int n = rem % N;
    dst[(long)mat * N * dstLdk + (long)n * dstLdk + dstKoff + k] =
        (f16)(scale * src[i]);
  }
}

// ---------------------------------------------------------------------------
// Core WMMA tile: one wave computes a 16x64 tile, acc += A(16xK) * B(Kx64)
// A is row-major [16][lda] f16 (tile base already applied)
// Bt is transposed weights [64][ldb] f16: Bt[n][k] = W[k][n] (tile base applied)
// Fragment k order: contiguous k = 16*half + s for both A and B (wave32).
// ---------------------------------------------------------------------------
template <int K>
__device__ static inline void wmma_tile_16x64(const f16* __restrict__ A, int lda,
                                              const f16* __restrict__ Bt, int ldb,
                                              v8f acc[4]) {
  const int lane = threadIdx.x & 31;
  const int l16  = lane & 15;
  const int half = lane >> 4;
  const f16* arow = A + (long)l16 * lda;
#pragma unroll
  for (int kk = 0; kk < K; kk += 32) {
    // A 16x32 f16 fragment: lane holds M = l16, K = 16*half + 0..15
    v16h a = *(const v16h*)(arow + kk + half * 16);
#pragma unroll
    for (int nt = 0; nt < 4; nt++) {
      // B 32x16 f16 fragment: lane holds N = nt*16 + l16, K = 16*half + 0..15
      v16h b = *(const v16h*)(Bt + (long)(nt * 16 + l16) * ldb + kk + half * 16);
      acc[nt] = __builtin_amdgcn_wmma_f32_16x16x32_f16(
          false, a, false, b, (short)0, acc[nt], false, false);
    }
  }
}

// Decode wave -> (mbase, nbase). 1024 m-tiles x 4 n-groups per matrix,
// 8 waves per 256-thread block, 512 blocks per matrix.
__device__ static inline void tile_coords(int& mbase, int& nbase) {
  int wave  = threadIdx.x >> 5;
  int gtile = blockIdx.x * 8 + wave;
  mbase = (gtile >> 2) * 16;
  nbase = (gtile & 3) * 64;
}

// ---------------------------------------------------------------------------
// GEMM 1: in_proj. z = blockIdx.y in 0..11 (k = z>>2, e = z&3)
// ir = gamma*(x@W+b) ; ii = gamma*(...) ; o = (...)
// ---------------------------------------------------------------------------
__global__ void __launch_bounds__(256)
gemm_inproj_kernel(const f16* __restrict__ xh, const f16* __restrict__ wt,
                   const float* __restrict__ ipb, const float* __restrict__ gmb,
                   float* __restrict__ ir, float* __restrict__ ii,
                   float* __restrict__ obuf) {
  int z = blockIdx.y;
  int mbase, nbase; tile_coords(mbase, nbase);
  const f16* A  = xh + (long)mbase * D_;
  const f16* Bt = wt + (long)z * D_ * D_ + (long)nbase * D_;
  v8f acc[4] = {};
  wmma_tile_16x64<D_>(A, D_, Bt, D_, acc);

  int k = z >> 2, e = z & 3;
  int lane = threadIdx.x & 31, l16 = lane & 15, half = lane >> 4;
#pragma unroll
  for (int nt = 0; nt < 4; nt++) {
    int n = nbase + nt * 16 + l16;
    float bia = ipb[z * D_ + n];
    float gm  = gmb[e * D_ + n];
#pragma unroll
    for (int r = 0; r < 8; r++) {
      int m = mbase + r + 8 * half;
      long idx = (long)e * MT * D_ + (long)m * D_ + n;
      float v = acc[nt][r] + bia;
      if (k == 0)       ir[idx]   = gm * v;
      else if (k == 1)  ii[idx]   = gm * v;
      else              obuf[idx] = v;
    }
  }
}

// ---------------------------------------------------------------------------
// LRU scan: 16384 independent channels (e,b,d), serial over T.
// Writes f16 [out_r | out_i] rows of width 512 (A matrix for the mid GEMM)
// and the packed final hidden state.
// ---------------------------------------------------------------------------
__global__ void __launch_bounds__(256)
lru_scan_kernel(const float* __restrict__ ir, const float* __restrict__ ii,
                const float* __restrict__ frb, const float* __restrict__ fib,
                f16* __restrict__ Amid, float* __restrict__ hidden) {
  int idx = blockIdx.x * blockDim.x + threadIdx.x;   // 0..16383
  int d = idx & 255;
  int b = (idx >> 8) & 15;
  int e = idx >> 12;
  float fr = frb[e * D_ + d];
  float fi = fib[e * D_ + d];
  long ibase = (long)e * MT * D_ + (long)b * T_ * D_ + d;
  long abase = (long)e * MT * 512 + (long)b * T_ * 512 + d;
  float hr = 0.f, hi = 0.f;
  for (int t = 0; t < T_; t++) {
    float ur = ir[ibase + (long)t * D_];
    float ui = ii[ibase + (long)t * D_];
    float hr2 = fr * hr - fi * hi + ur;
    float hi2 = fr * hi + fi * hr + ui;
    hr = hr2; hi = hi2;
    Amid[abase + (long)t * 512]       = (f16)hr;
    Amid[abase + (long)t * 512 + 256] = (f16)hi;
  }
  hidden[b * 2 * E_ * D_ + e * D_ + d]           = hr;
  hidden[b * 2 * E_ * D_ + E_ * D_ + e * D_ + d] = hi;
}

// ---------------------------------------------------------------------------
// GEMM 2: mid. K=512 with W = [mid_w0 ; -mid_w1] pre-concatenated.
// out = A@W + (b0-b1) + o ; write f32 (for residual) and f16 (ff1 input)
// ---------------------------------------------------------------------------
__global__ void __launch_bounds__(256)
gemm_mid_kernel(const f16* __restrict__ Amid, const f16* __restrict__ wtm,
                const float* __restrict__ midb, const float* __restrict__ obuf,
                float* __restrict__ outf, f16* __restrict__ outh) {
  int e = blockIdx.y;
  int mbase, nbase; tile_coords(mbase, nbase);
  const f16* A  = Amid + (long)e * MT * 512 + (long)mbase * 512;
  const f16* Bt = wtm  + (long)e * D_ * 512 + (long)nbase * 512;
  v8f acc[4] = {};
  wmma_tile_16x64<512>(A, 512, Bt, 512, acc);

  int lane = threadIdx.x & 31, l16 = lane & 15, half = lane >> 4;
#pragma unroll
  for (int nt = 0; nt < 4; nt++) {
    int n = nbase + nt * 16 + l16;
    float bia = midb[e * D_ + n] - midb[E_ * D_ + e * D_ + n];
#pragma unroll
    for (int r = 0; r < 8; r++) {
      int m = mbase + r + 8 * half;
      long idx = (long)e * MT * D_ + (long)m * D_ + n;
      float v = acc[nt][r] + bia + obuf[idx];
      outf[idx] = v;
      outh[idx] = (f16)v;
    }
  }
}

// ---------------------------------------------------------------------------
// GEMM 3: ff1 + exact GELU
// ---------------------------------------------------------------------------
__global__ void __launch_bounds__(256)
gemm_ff1_kernel(const f16* __restrict__ outh, const f16* __restrict__ wt1,
                const float* __restrict__ fb1, f16* __restrict__ h1h) {
  int e = blockIdx.y;
  int mbase, nbase; tile_coords(mbase, nbase);
  const f16* A  = outh + (long)e * MT * D_ + (long)mbase * D_;
  const f16* Bt = wt1  + (long)e * D_ * D_ + (long)nbase * D_;
  v8f acc[4] = {};
  wmma_tile_16x64<D_>(A, D_, Bt, D_, acc);

  int lane = threadIdx.x & 31, l16 = lane & 15, half = lane >> 4;
#pragma unroll
  for (int nt = 0; nt < 4; nt++) {
    int n = nbase + nt * 16 + l16;
    float bia = fb1[e * D_ + n];
#pragma unroll
    for (int r = 0; r < 8; r++) {
      int m = mbase + r + 8 * half;
      float v = acc[nt][r] + bia;
      float g = 0.5f * v * (1.0f + erff(v * 0.70710678118654752f));
      h1h[(long)e * MT * D_ + (long)m * D_ + n] = (f16)g;
    }
  }
}

// ---------------------------------------------------------------------------
// GEMM 4: ff2 + residual -> h2 (f32)
// ---------------------------------------------------------------------------
__global__ void __launch_bounds__(256)
gemm_ff2_kernel(const f16* __restrict__ h1h, const f16* __restrict__ wt2,
                const float* __restrict__ fb2, const float* __restrict__ outf,
                float* __restrict__ h2) {
  int e = blockIdx.y;
  int mbase, nbase; tile_coords(mbase, nbase);
  const f16* A  = h1h + (long)e * MT * D_ + (long)mbase * D_;
  const f16* Bt = wt2 + (long)e * D_ * D_ + (long)nbase * D_;
  v8f acc[4] = {};
  wmma_tile_16x64<D_>(A, D_, Bt, D_, acc);

  int lane = threadIdx.x & 31, l16 = lane & 15, half = lane >> 4;
#pragma unroll
  for (int nt = 0; nt < 4; nt++) {
    int n = nbase + nt * 16 + l16;
    float bia = fb2[e * D_ + n];
#pragma unroll
    for (int r = 0; r < 8; r++) {
      int m = mbase + r + 8 * half;
      long idx = (long)e * MT * D_ + (long)m * D_ + n;
      h2[idx] = acc[nt][r] + bia + outf[idx];
    }
  }
}

// ---------------------------------------------------------------------------
// LayerNorm over (E, D) jointly, per (b,t). One 256-thread block per m.
// ---------------------------------------------------------------------------
__global__ void __launch_bounds__(256)
layernorm_kernel(const float* __restrict__ h2, const float* __restrict__ lnw,
                 const float* __restrict__ lnb, float* __restrict__ out) {
  int m = blockIdx.x;          // b*T + t
  int d = threadIdx.x;         // 0..255
  float v[E_];
  float s = 0.f, sq = 0.f;
#pragma unroll
  for (int e = 0; e < E_; e++) {
    v[e] = h2[(long)e * MT * D_ + (long)m * D_ + d];
    s  += v[e];
    sq += v[e] * v[e];
  }
  __shared__ float rs[256], rq[256];
  rs[d] = s; rq[d] = sq;
  __syncthreads();
  for (int off = 128; off > 0; off >>= 1) {
    if (d < off) { rs[d] += rs[d + off]; rq[d] += rq[d + off]; }
    __syncthreads();
  }
  float mean = rs[0] * (1.0f / (E_ * D_));
  float var  = rq[0] * (1.0f / (E_ * D_)) - mean * mean;
  float rstd = rsqrtf(var + LN_EPS);
#pragma unroll
  for (int e = 0; e < E_; e++) {
    out[(long)e * MT * D_ + (long)m * D_ + d] =
        (v[e] - mean) * rstd * lnw[e * D_ + d] + lnb[e * D_ + d];
  }
}

// ---------------------------------------------------------------------------
// Host launcher
// ---------------------------------------------------------------------------
extern "C" void kernel_launch(void* const* d_in, const int* in_sizes, int n_in,
                              void* d_out, int out_size, void* d_ws, size_t ws_size,
                              hipStream_t stream) {
  const float* x    = (const float*)d_in[0];
  const float* ipw  = (const float*)d_in[1];
  const float* ipb  = (const float*)d_in[2];
  const float* plog = (const float*)d_in[3];
  const float* midw = (const float*)d_in[4];
  const float* midb = (const float*)d_in[5];
  const float* ffw1 = (const float*)d_in[6];
  const float* ffb1 = (const float*)d_in[7];
  const float* ffw2 = (const float*)d_in[8];
  const float* ffb2 = (const float*)d_in[9];
  const float* lnw  = (const float*)d_in[10];
  const float* lnb  = (const float*)d_in[11];

  float* out_ln = (float*)d_out;
  float* hidden = (float*)d_out + (long)E_ * MT * D_;

  char* ws = (char*)d_ws;
  size_t off = 0;
  f16* xh     = (f16*)(ws + off); off += (size_t)MT * D_ * 2;        // 8 MB
  f16* wt_in  = (f16*)(ws + off); off += 12ull * D_ * D_ * 2;        // 1.5 MB
  f16* wt_mid = (f16*)(ws + off); off += 4ull * D_ * 512 * 2;        // 1 MB
  f16* wt_f1  = (f16*)(ws + off); off += 4ull * D_ * D_ * 2;         // 0.5 MB
  f16* wt_f2  = (f16*)(ws + off); off += 4ull * D_ * D_ * 2;         // 0.5 MB
  float* frb  = (float*)(ws + off); off += E_ * D_ * 4;
  float* fib  = (float*)(ws + off); off += E_ * D_ * 4;
  float* gmb  = (float*)(ws + off); off += E_ * D_ * 4;
  off = (off + 255) & ~(size_t)255;

  size_t big = (size_t)E_ * MT * D_ * 4;                             // 64 MB
  size_t irOff = off;           float* ir   = (float*)(ws + irOff); off += big;
  size_t iiOff = off;           float* ii   = (float*)(ws + iiOff); off += big;
  size_t oOff  = off;           float* obuf = (float*)(ws + oOff);  off += big;
  f16* Amid = (f16*)(ws + off); off += (size_t)E_ * MT * 512 * 2;    // 64 MB

  // Aliases over dead buffers (lifetimes do not overlap):
  float* outf = (float*)(ws + irOff);                   // reuse ir (after scan)
  f16*   outh = (f16*)(ws + iiOff);                     // reuse ii lower 32 MB
  f16*   h1h  = (f16*)(ws + iiOff + big / 2);           // reuse ii upper 32 MB
  float* h2   = (float*)(ws + oOff);                    // reuse obuf (after mid)

  // --- prep ---
  prep_params_kernel<<<4, 256, 0, stream>>>(plog, frb, fib, gmb);
  cvt_f32_f16_kernel<<<4096, 256, 0, stream>>>(x, xh, MT * D_);
  transpose_cvt_kernel<<<1024, 256, 0, stream>>>(ipw, wt_in, D_, D_, 12, D_, 0, 1.0f);
  transpose_cvt_kernel<<<1024, 256, 0, stream>>>(midw, wt_mid, D_, D_, 4, 512, 0, 1.0f);
  transpose_cvt_kernel<<<1024, 256, 0, stream>>>(midw + 4ull * D_ * D_, wt_mid,
                                                 D_, D_, 4, 512, 256, -1.0f);
  transpose_cvt_kernel<<<1024, 256, 0, stream>>>(ffw1, wt_f1, D_, D_, 4, D_, 0, 1.0f);
  transpose_cvt_kernel<<<1024, 256, 0, stream>>>(ffw2, wt_f2, D_, D_, 4, D_, 0, 1.0f);

  // --- pipeline ---
  gemm_inproj_kernel<<<dim3(512, 12), 256, 0, stream>>>(xh, wt_in, ipb, gmb,
                                                        ir, ii, obuf);
  lru_scan_kernel<<<64, 256, 0, stream>>>(ir, ii, frb, fib, Amid, hidden);
  gemm_mid_kernel<<<dim3(512, 4), 256, 0, stream>>>(Amid, wt_mid, midb, obuf,
                                                    outf, outh);
  gemm_ff1_kernel<<<dim3(512, 4), 256, 0, stream>>>(outh, wt_f1, ffb1, h1h);
  gemm_ff2_kernel<<<dim3(512, 4), 256, 0, stream>>>(h1h, wt_f2, ffb2, outf, h2);
  layernorm_kernel<<<16384, 256, 0, stream>>>(h2, lnw, lnb, out_ln);
}
